// MultiHeadSelfAttention_52536039964959
// MI455X (gfx1250) — compile-verified, matrix operand order checked by reading, not run on hardware
//
#include <hip/hip_runtime.h>
#include <hip/hip_bf16.h>

#define DMODEL 1024
#define NHEAD  16
#define DKH    64

typedef __attribute__((ext_vector_type(16))) __bf16 v16bf;
typedef __attribute__((ext_vector_type(8)))  __bf16 v8bf;
typedef __attribute__((ext_vector_type(8)))  float  v8f;
typedef __attribute__((ext_vector_type(4)))  float  v4f;
typedef __attribute__((ext_vector_type(4)))  unsigned int u32x4;
typedef __attribute__((ext_vector_type(8)))  int i32x8;
typedef __attribute__((ext_vector_type(4)))  int i32x4;

static __device__ __forceinline__ v8f wmma_bf16(v16bf a, v16bf b, v8f c) {
  return __builtin_amdgcn_wmma_f32_16x16x32_bf16(false, a, false, b, (short)0, c, false, false);
}

// 16x32 bf16 fragment load (A or B; identical per-lane pattern when the source
// is row-major over K).  lane<16 -> K {0..7,16..23}, lane>=16 -> {8..15,24..31}
// per CDNA5 ISA 7.12.2.
static __device__ __forceinline__ v16bf load_frag16(const __bf16* t, int pitch, int rbase, int kbase) {
  const int lane = threadIdx.x & 31;
  const __bf16* p = t + (size_t)(rbase + (lane & 15)) * pitch + kbase + ((lane >> 4) << 3);
  v8bf lo = *(const v8bf*)p;
  v8bf hi = *(const v8bf*)(p + 16);
  return __builtin_shufflevector(lo, hi, 0,1,2,3,4,5,6,7,8,9,10,11,12,13,14,15);
}

static __device__ __forceinline__ v8bf cvt_f32x8_bf16(const float* p) {
  v4f a = *(const v4f*)p;
  v4f b = *(const v4f*)(p + 4);
  v8bf r;
#pragma unroll
  for (int i = 0; i < 4; ++i) { r[i] = (__bf16)a[i]; r[i + 4] = (__bf16)b[i]; }
  return r;
}

// ---------------------------------------------------------------------------
// TDM: issue a 2-D bf16 tile DMA (global -> LDS) via a Tensor DMA Descriptor.
// D# group0: count=1, lds_addr, 57-bit global addr, type=2 (ISA 8.3).
// D# group1: data_size=1 (2B), optional LDS pad, tensor/tile dims, dim0 stride
// (ISA 8.4).  Groups 2/3 zero (<=2D tensor).  Tracked by TENSORcnt.
// ---------------------------------------------------------------------------
static __device__ __forceinline__ void tdm_load_2d_bf16(
    const void* gaddr, unsigned lds_off,
    unsigned tensor_d0, unsigned tensor_d1,
    unsigned tile_d0, unsigned tile_d1,
    unsigned long long stride0_elems,
    unsigned pad_interval_code, unsigned pad_amount_code)
{
  unsigned long long ga = (unsigned long long)gaddr;
  u32x4 g0;
  g0[0] = 1u;                                                 // count=1
  g0[1] = lds_off;                                            // lds_addr (bytes)
  g0[2] = (unsigned)(ga & 0xffffffffu);                       // global_addr lo
  g0[3] = (unsigned)((ga >> 32) & 0x01ffffffu) | (2u << 30);  // addr hi | type=2

  i32x8 g1;
  unsigned dw0 = (1u << 16)                                   // data_size = 2B
               | (1u << 20)                                   // pad_enable
               | (pad_interval_code << 22)
               | (pad_amount_code << 25);
  g1[0] = (int)dw0;
  g1[1] = (int)((tensor_d0 & 0xffffu) << 16);                 // dim0[15:0]
  g1[2] = (int)((tensor_d0 >> 16) | ((tensor_d1 & 0xffffu) << 16));
  g1[3] = (int)((tensor_d1 >> 16) | (tile_d0 << 16));         // tile_dim0
  g1[4] = (int)(tile_d1 & 0xffffu);                           // tile_dim1; tile_dim2=0
  g1[5] = (int)(unsigned)(stride0_elems & 0xffffffffull);     // dim0_stride lo32
  g1[6] = (int)(unsigned)((stride0_elems >> 32) & 0xffffull); // dim0_stride hi16
  g1[7] = 0;
  i32x4 z = {0, 0, 0, 0};
#if defined(__clang_major__) && __clang_major__ >= 23
  i32x8 z8 = {0, 0, 0, 0, 0, 0, 0, 0};
  __builtin_amdgcn_tensor_load_to_lds(g0, g1, z, z, z8, 0);
#else
  __builtin_amdgcn_tensor_load_to_lds(g0, g1, z, z, 0);
#endif
}

// ---------------------------------------------------------------------------
// Kernel 1: fused Q/K/V projections.  C = X . W^T + bias.  128x64 tile per
// block, 2x2 register blocking per wave (4 WMMA per K-step).  Output bf16:
// q,k -> [B,H,S,64]; v -> transposed [B,H,64,S].
// ---------------------------------------------------------------------------
__global__ __launch_bounds__(256) void qkv_proj_kernel(
    const float* __restrict__ Qin, const float* __restrict__ Kin, const float* __restrict__ Vin,
    const float* __restrict__ Wq,  const float* __restrict__ Wk,  const float* __restrict__ Wv,
    const float* __restrict__ Bq,  const float* __restrict__ Bk,  const float* __restrict__ Bv,
    __bf16* __restrict__ qo, __bf16* __restrict__ ko, __bf16* __restrict__ vto, int S)
{
  const int mode = blockIdx.z;
  const float* X  = (mode == 0) ? Qin : (mode == 1) ? Kin : Vin;
  const float* W  = (mode == 0) ? Wq  : (mode == 1) ? Wk  : Wv;
  const float* Bi = (mode == 0) ? Bq  : (mode == 1) ? Bk  : Bv;

  __shared__ __bf16 Xs[128 * 40];
  __shared__ __bf16 Ws[64 * 40];
  const int P = 40;

  const int t    = threadIdx.x;
  const int lane = t & 31;
  const int wave = t >> 5;
  const int half = lane >> 4;
  const int l16  = lane & 15;

  const int Mb = blockIdx.x * 128;
  const int Nb = blockIdx.y * 64;

  const int mg = (wave & 3) * 32;   // wave's 32-row group
  const int ng = (wave >> 2) * 32;  // wave's 32-col group

  v8f acc00 = {}, acc01 = {}, acc10 = {}, acc11 = {};

  const int xr = t >> 1, xc = (t & 1) * 16;   // X stage: 128 rows x 32 cols
  const int wr = t >> 2, wc = (t & 3) * 8;    // W stage:  64 rows x 32 cols
  const float* gx = X + (size_t)(Mb + xr) * DMODEL + xc;
  const float* gw = W + (size_t)(Nb + wr) * DMODEL + wc;

  for (int k0 = 0; k0 < DMODEL; k0 += 32) {
    *(v8bf*)&Xs[xr * P + xc]     = cvt_f32x8_bf16(gx + k0);
    *(v8bf*)&Xs[xr * P + xc + 8] = cvt_f32x8_bf16(gx + k0 + 8);
    *(v8bf*)&Ws[wr * P + wc]     = cvt_f32x8_bf16(gw + k0);
    if (k0 + 32 < DMODEL) {
      __builtin_prefetch(gx + k0 + 32, 0, 3);
      __builtin_prefetch(gw + k0 + 32, 0, 3);
    }
    __syncthreads();
    v16bf a0 = load_frag16(Xs, P, mg,      0);
    v16bf a1 = load_frag16(Xs, P, mg + 16, 0);
    v16bf b0 = load_frag16(Ws, P, ng,      0);
    v16bf b1 = load_frag16(Ws, P, ng + 16, 0);
    acc00 = wmma_bf16(a0, b0, acc00);
    acc01 = wmma_bf16(a0, b1, acc01);
    acc10 = wmma_bf16(a1, b0, acc10);
    acc11 = wmma_bf16(a1, b1, acc11);
    __syncthreads();
  }

  // Epilogue.  C layout: lane holds col n; VGPR v holds row M = v + 8*half.
#pragma unroll
  for (int mt = 0; mt < 2; ++mt) {
#pragma unroll
    for (int nt = 0; nt < 2; ++nt) {
      v8f acc = (mt == 0) ? (nt == 0 ? acc00 : acc01) : (nt == 0 ? acc10 : acc11);
      const int n = Nb + ng + nt * 16 + l16;
      const float bias = Bi[n];
      const int h = n >> 6;
      const int d = n & 63;
      const int mb   = Mb + mg + mt * 16 + half * 8;
      const int bidx = mb / S;
      const int srow = mb % S;
      if (mode != 2) {
        __bf16* o = (mode == 0) ? qo : ko;
#pragma unroll
        for (int v = 0; v < 8; ++v) {
          size_t addr = (((size_t)(bidx * NHEAD + h) * S) + (srow + v)) * DKH + d;
          o[addr] = (__bf16)(acc[v] + bias);
        }
      } else {
        v8bf ov;
#pragma unroll
        for (int v = 0; v < 8; ++v) ov[v] = (__bf16)(acc[v] + bias);
        size_t addr = ((size_t)(bidx * NHEAD + h) * DKH + d) * S + srow;  // 8 consecutive s
        *(v8bf*)&vto[addr] = ov;
      }
    }
  }
}

// ---------------------------------------------------------------------------
// Kernel 2: flash attention.  Block = 128 queries of one (b,h).  K / V^T tiles
// are staged into double-buffered LDS by the Tensor Data Mover (TDM pad fields
// recreate the fragment-loader's pitch padding), overlapped with compute via
// TENSORcnt.  Scores + P.V via WMMA, online softmax in fp32.
// ---------------------------------------------------------------------------
__global__ __launch_bounds__(256) void flash_attn_kernel(
    const __bf16* __restrict__ q, const __bf16* __restrict__ k,
    const __bf16* __restrict__ vt, __bf16* __restrict__ ctx, int S)
{
  const int b  = blockIdx.z, h = blockIdx.y;
  const int qb = blockIdx.x * 128;
  const int t = threadIdx.x, lane = t & 31, wave = t >> 5;
  const int half = lane >> 4, l16 = lane & 15;

  __shared__ __bf16 Ks[2][32 * 72];    // [buf][key][d]   (pitch 72 via TDM pad)
  __shared__ __bf16 VTs[2][64 * 40];   // [buf][d][key]   (pitch 40 via TDM pad)
  __shared__ __bf16 Ps[8][16 * 40];    // per-wave P tile [qrow][key]
  const int KP = 72, VP = 40, PP = 40;

  const size_t bh = (size_t)(b * NHEAD + h);
  const __bf16* qp = q  + bh * S * DKH;
  const __bf16* kp = k  + bh * S * DKH;
  const __bf16* vp = vt + bh * DKH * S;

  // Q fragments held in registers for the whole key loop (K = 0..31, 32..63).
  v16bf qf0, qf1;
  {
    const __bf16* p = qp + (size_t)(qb + wave * 16 + l16) * DKH + (half << 3);
    v8bf a0 = *(const v8bf*)p;
    v8bf a1 = *(const v8bf*)(p + 16);
    v8bf a2 = *(const v8bf*)(p + 32);
    v8bf a3 = *(const v8bf*)(p + 48);
    qf0 = __builtin_shufflevector(a0, a1, 0,1,2,3,4,5,6,7,8,9,10,11,12,13,14,15);
    qf1 = __builtin_shufflevector(a2, a3, 0,1,2,3,4,5,6,7,8,9,10,11,12,13,14,15);
  }

  float mrow[8], lsum[8];
#pragma unroll
  for (int i = 0; i < 8; ++i) { mrow[i] = -3.0e38f; lsum[i] = 0.f; }
  v8f O0 = {}, O1 = {}, O2 = {}, O3 = {};
  const float scale = 0.125f;  // 1/sqrt(64)

  // Prologue: DMA first K (32x64) and V^T (64x32) tiles into buffer 0.
  if (wave == 0) {
    tdm_load_2d_bf16(kp, (unsigned)(size_t)&Ks[0][0], 64, 32, 64, 32, 64, 4, 3);
    tdm_load_2d_bf16(vp, (unsigned)(size_t)&VTs[0][0], 32, 64, 32, 64,
                     (unsigned long long)S, 3, 3);
  }

  int cur = 0;
  for (int kb = 0; kb < S; kb += 32, cur ^= 1) {
    if (wave == 0) {
      if (kb + 32 < S) {
        // Issue next tile pair into the alternate buffer, then wait for the
        // current pair (TDM ops complete in order -> tensorcnt<=2).
        tdm_load_2d_bf16(kp + (size_t)(kb + 32) * DKH,
                         (unsigned)(size_t)&Ks[cur ^ 1][0], 64, 32, 64, 32, 64, 4, 3);
        tdm_load_2d_bf16(vp + (kb + 32),
                         (unsigned)(size_t)&VTs[cur ^ 1][0], 32, 64, 32, 64,
                         (unsigned long long)S, 3, 3);
        __builtin_amdgcn_s_wait_tensorcnt(2);
      } else {
        __builtin_amdgcn_s_wait_tensorcnt(0);
      }
    }
    __syncthreads();
    const __bf16* Kc = &Ks[cur][0];
    const __bf16* Vc = &VTs[cur][0];

    // Scores: 16 queries x 32 keys (two 16-key N tiles, K=64 over d).
    v8f s0 = {}, s1 = {};
    s0 = wmma_bf16(qf0, load_frag16(Kc, KP,  0,  0), s0);
    s0 = wmma_bf16(qf1, load_frag16(Kc, KP,  0, 32), s0);
    s1 = wmma_bf16(qf0, load_frag16(Kc, KP, 16,  0), s1);
    s1 = wmma_bf16(qf1, load_frag16(Kc, KP, 16, 32), s1);

    // Online softmax.  Score/O row M = v + 8*half, so the 16-lane-half
    // reductions and the O rescale are layout-consistent.
    __bf16* pw = &Ps[wave][0];
#pragma unroll
    for (int v = 0; v < 8; ++v) {
      float a = s0[v] * scale;
      float c = s1[v] * scale;
      float mx = fmaxf(a, c);
      mx = fmaxf(mx, __shfl_xor(mx, 1, 16));
      mx = fmaxf(mx, __shfl_xor(mx, 2, 16));
      mx = fmaxf(mx, __shfl_xor(mx, 4, 16));
      mx = fmaxf(mx, __shfl_xor(mx, 8, 16));
      float mnew = fmaxf(mrow[v], mx);
      float corr = __expf(mrow[v] - mnew);
      float p0 = __expf(a - mnew);
      float p1 = __expf(c - mnew);
      float rs = p0 + p1;
      rs += __shfl_xor(rs, 1, 16);
      rs += __shfl_xor(rs, 2, 16);
      rs += __shfl_xor(rs, 4, 16);
      rs += __shfl_xor(rs, 8, 16);
      lsum[v] = lsum[v] * corr + rs;
      mrow[v] = mnew;
      O0[v] *= corr; O1[v] *= corr; O2[v] *= corr; O3[v] *= corr;
      const int pidx = (v + half * 8) * PP + l16;  // C-layout -> [row][key]
      pw[pidx]      = (__bf16)p0;
      pw[pidx + 16] = (__bf16)p1;
    }
    // Same-wave cross-lane LDS handoff: wait for all lanes' P stores.
    asm volatile("s_wait_dscnt 0" ::: "memory");

    // P.V : A = P (16 queries x 32 keys), B = V from V^T LDS tile.
    v16bf pa = load_frag16(pw, PP, 0, 0);
    O0 = wmma_bf16(pa, load_frag16(Vc, VP,  0, 0), O0);
    O1 = wmma_bf16(pa, load_frag16(Vc, VP, 16, 0), O1);
    O2 = wmma_bf16(pa, load_frag16(Vc, VP, 32, 0), O2);
    O3 = wmma_bf16(pa, load_frag16(Vc, VP, 48, 0), O3);
    __syncthreads();
  }

  // Normalize and write context in [B,S,DMODEL] layout (ready as GEMM A).
#pragma unroll
  for (int dt = 0; dt < 4; ++dt) {
    v8f O = (dt == 0) ? O0 : (dt == 1) ? O1 : (dt == 2) ? O2 : O3;
#pragma unroll
    for (int v = 0; v < 8; ++v) {
      float val = O[v] / lsum[v];
      int srow = qb + wave * 16 + half * 8 + v;
      ctx[((size_t)(b * S + srow)) * DMODEL + h * DKH + dt * 16 + l16] = (__bf16)val;
    }
  }
}

// ---------------------------------------------------------------------------
// Kernel 3: output projection.  out = ctx . Wo^T + bo (ctx bf16, out fp32),
// same 128x64 / 2x2-blocked scheme as kernel 1.
// ---------------------------------------------------------------------------
__global__ __launch_bounds__(256) void out_proj_kernel(
    const __bf16* __restrict__ Xc, const float* __restrict__ Wo,
    const float* __restrict__ Bo, float* __restrict__ out)
{
  __shared__ __bf16 Xs[128 * 40];
  __shared__ __bf16 Ws[64 * 40];
  const int P = 40;

  const int t    = threadIdx.x;
  const int lane = t & 31;
  const int wave = t >> 5;
  const int half = lane >> 4;
  const int l16  = lane & 15;

  const int Mb = blockIdx.x * 128;
  const int Nb = blockIdx.y * 64;

  const int mg = (wave & 3) * 32;
  const int ng = (wave >> 2) * 32;

  v8f acc00 = {}, acc01 = {}, acc10 = {}, acc11 = {};

  const int xr = t >> 1, xc = (t & 1) * 16;
  const int wr = t >> 2, wc = (t & 3) * 8;
  const __bf16* gx = Xc + (size_t)(Mb + xr) * DMODEL + xc;
  const float*  gw = Wo + (size_t)(Nb + wr) * DMODEL + wc;

  for (int k0 = 0; k0 < DMODEL; k0 += 32) {
    *(v8bf*)&Xs[xr * P + xc]     = *(const v8bf*)(gx + k0);
    *(v8bf*)&Xs[xr * P + xc + 8] = *(const v8bf*)(gx + k0 + 8);
    *(v8bf*)&Ws[wr * P + wc]     = cvt_f32x8_bf16(gw + k0);
    if (k0 + 32 < DMODEL) {
      __builtin_prefetch(gx + k0 + 32, 0, 3);
      __builtin_prefetch(gw + k0 + 32, 0, 3);
    }
    __syncthreads();
    v16bf a0 = load_frag16(Xs, P, mg,      0);
    v16bf a1 = load_frag16(Xs, P, mg + 16, 0);
    v16bf b0 = load_frag16(Ws, P, ng,      0);
    v16bf b1 = load_frag16(Ws, P, ng + 16, 0);
    acc00 = wmma_bf16(a0, b0, acc00);
    acc01 = wmma_bf16(a0, b1, acc01);
    acc10 = wmma_bf16(a1, b0, acc10);
    acc11 = wmma_bf16(a1, b1, acc11);
    __syncthreads();
  }

#pragma unroll
  for (int mt = 0; mt < 2; ++mt) {
#pragma unroll
    for (int nt = 0; nt < 2; ++nt) {
      v8f acc = (mt == 0) ? (nt == 0 ? acc00 : acc01) : (nt == 0 ? acc10 : acc11);
      const int n = Nb + ng + nt * 16 + l16;
      const float bias = Bo[n];
      const int mb = Mb + mg + mt * 16 + half * 8;
#pragma unroll
      for (int v = 0; v < 8; ++v) {
        out[(size_t)(mb + v) * DMODEL + n] = acc[v] + bias;
      }
    }
  }
}

extern "C" void kernel_launch(void* const* d_in, const int* in_sizes, int n_in,
                              void* d_out, int out_size, void* d_ws, size_t ws_size,
                              hipStream_t stream) {
  const float* Q  = (const float*)d_in[0];
  const float* K  = (const float*)d_in[1];
  const float* V  = (const float*)d_in[2];
  const float* wq = (const float*)d_in[3];
  const float* bq = (const float*)d_in[4];
  const float* wk = (const float*)d_in[5];
  const float* bk = (const float*)d_in[6];
  const float* wv = (const float*)d_in[7];
  const float* bv = (const float*)d_in[8];
  const float* wo = (const float*)d_in[9];
  const float* bo = (const float*)d_in[10];

  const int rows = in_sizes[0] / DMODEL;  // B * S
  const int S = 2048;
  const int B = rows / S;

  char* ws = (char*)d_ws;
  const size_t seg = (size_t)rows * DMODEL * sizeof(__bf16);
  __bf16* qws   = (__bf16*)ws;               // [B,H,S,64]
  __bf16* kws   = (__bf16*)(ws + seg);       // [B,H,S,64]
  __bf16* vtws  = (__bf16*)(ws + 2 * seg);   // [B,H,64,S]
  __bf16* ctxws = (__bf16*)(ws + 3 * seg);   // [B,S,DMODEL]

  dim3 gProj(rows / 128, DMODEL / 64, 3);
  qkv_proj_kernel<<<gProj, 256, 0, stream>>>(Q, K, V, wq, wk, wv, bq, bk, bv,
                                             qws, kws, vtws, S);

  dim3 gAttn(S / 128, NHEAD, B);
  flash_attn_kernel<<<gAttn, 256, 0, stream>>>(qws, kws, vtws, ctxws, S);

  dim3 gOut(rows / 128, DMODEL / 64, 1);
  out_proj_kernel<<<gOut, 256, 0, stream>>>(ctxws, wo, bo, (float*)d_out);
}
